// LSTMModel_16750372454500
// MI455X (gfx1250) — matrix operand order repaired and checked by reading, not compiled, not run on hardware
//
#include <hip/hip_runtime.h>

typedef __bf16 v16bf __attribute__((ext_vector_type(16)));
typedef float  v8f   __attribute__((ext_vector_type(8)));
typedef unsigned int u32x4 __attribute__((ext_vector_type(4)));
typedef int          i32x8 __attribute__((ext_vector_type(8)));
typedef int          i32x4 __attribute__((ext_vector_type(4)));

#define B_   128
#define T_   1024
#define Y_   32
#define H_   256

#if __has_builtin(__builtin_amdgcn_tensor_load_to_lds)
#define USE_TDM 1
#else
#define USE_TDM 0
#endif

// ---------- helpers ----------
__device__ __forceinline__ unsigned short f2bf(float f) {
    unsigned int u = __float_as_uint(f);
    u += 0x7FFFu + ((u >> 16) & 1u);          // round-to-nearest-even
    return (unsigned short)(u >> 16);
}
__device__ __forceinline__ float sigmoidf_(float x) {
    return 1.0f / (1.0f + __expf(-x));
}

// ---------------------------------------------------------------------------
// Pack W_hh -> bf16 WMMA B-fragments.
// Fragment (ht, g, kt): B is 32(K) x 16(N); lanes 0-15 hold K=kt*32+0..7,16..23,
// lanes 16-31 hold K=kt*32+8..15,24..31. N = ht*16+(lane&15).
// ---------------------------------------------------------------------------
__global__ void pack_whh(const float* __restrict__ Whh, unsigned short* __restrict__ out) {
    int i = blockIdx.x * blockDim.x + threadIdx.x;
    if (i >= 16 * 4 * 8 * 512) return;
    int idx  = i & 15;
    int lane = (i >> 4) & 31;
    int fi   = i >> 9;                // ((ht*4+g)*8+kt)
    int kt = fi & 7, g = (fi >> 3) & 3, ht = fi >> 5;
    int n = g * 256 + ht * 16 + (lane & 15);
    int k = kt * 32 + idx + ((idx >= 8) ? 8 : 0) + ((lane >= 16) ? 8 : 0);
    out[i] = f2bf(Whh[n * H_ + k]);
}

// W_ih -> bf16 B-fragments (single K-tile, K=32).
__global__ void pack_wih(const float* __restrict__ Wih, unsigned short* __restrict__ out) {
    int i = blockIdx.x * blockDim.x + threadIdx.x;
    if (i >= 4 * 16 * 512) return;
    int idx  = i & 15;
    int lane = (i >> 4) & 31;
    int fi   = i >> 9;                // g*16+ht
    int ht = fi & 15, g = fi >> 4;
    int n = g * 256 + ht * 16 + (lane & 15);
    int k = idx + ((idx >= 8) ? 8 : 0) + ((lane >= 16) ? 8 : 0);
    out[i] = f2bf(Wih[n * Y_ + k]);
}

// y_seq [B,T,32] -> bf16 A-fragments (16x32). One step = 8 fragments = 8 KB contiguous.
__global__ void pack_y(const float* __restrict__ y, unsigned short* __restrict__ out) {
    int i = blockIdx.x * blockDim.x + threadIdx.x;
    if (i >= T_ * 8 * 512) return;
    int idx  = i & 15;
    int lane = (i >> 4) & 31;
    int fi   = i >> 9;                // t*8+mt
    int mt = fi & 7, t = fi >> 3;
    int b = mt * 16 + (lane & 15);
    int k = idx + ((idx >= 8) ? 8 : 0) + ((lane >= 16) ? 8 : 0);
    out[i] = f2bf(y[(b * T_ + t) * Y_ + k]);
}

// ---------------------------------------------------------------------------
// Persistent LSTM: one workgroup, 512 threads = 16 wave32s (4/SIMD, 256-VGPR
// budget). Wave w owns hidden slice ht=w and all 8 batch tiles + 4 gates.
// Dynamic LDS (272 KB): h double-buffer in A-fragment layout (2x64 KB),
// c-state f32 (128 KB), TDM y staging (2x8 KB). One barrier per timestep.
// Next step's y fragments are DMA'd LDS-ward by the Tensor Data Mover while
// the current step computes (wave 0 issues, s_wait_tensorcnt before barrier).
// ---------------------------------------------------------------------------
__global__ __launch_bounds__(512, 1)
void lstm_persistent(const unsigned short* __restrict__ whhP,
                     const unsigned short* __restrict__ wihP,
                     const unsigned short* __restrict__ yP,
                     const float* __restrict__ b_ih,
                     const float* __restrict__ b_hh,
                     float* __restrict__ hFin) {
    extern __shared__ char smem[];
    unsigned short* hA = (unsigned short*)smem;            // [2][8mt][8kt][32][16] = 128 KB
    float*          cS = (float*)(smem + 2 * 65536);       // [128][256] f32     = 128 KB
    unsigned short* yS = (unsigned short*)(smem + 4 * 65536); // [2][8mt][32][16] = 16 KB

    const int tid  = threadIdx.x;
    const int lane = tid & 31;
    const int ht   = tid >> 5;        // wave id == h-tile

    for (int i = tid; i < 8 * 8 * 32 * 16; i += 512) hA[i] = 0;
    for (int i = tid; i < B_ * H_; i += 512) cS[i] = 0.f;

    // per-lane gate biases: N is lane-constant in the C/D layout
    const int hidx = ht * 16 + (lane & 15);
    float bias[4];
#pragma unroll
    for (int g = 0; g < 4; ++g) {
        int n = g * 256 + hidx;
        bias[g] = b_ih[n] + b_hh[n];
    }

    // W_ih B-fragments are step-invariant: pin them in VGPRs
    v16bf Bi[4];
#pragma unroll
    for (int g = 0; g < 4; ++g)
        Bi[g] = *(const v16bf*)(wihP + (((g * 16 + ht) * 32) + lane) * 16);

    // lane-constant pieces of the h -> A-fragment scatter address
    const int ktp  = hidx >> 5;
    const int ko   = hidx & 31;
    const int vv   = ((ko & 7) >> 1) + ((ko & 16) ? 4 : 0);
    const int xx   = ko & 1;
    const int hi16 = (ko >> 3) & 1;
    const int off  = (lane < 16) ? 0 : 8;

#if USE_TDM
    const unsigned ldsBaseY = (unsigned)(unsigned long long)(void*)yS;
    // Preload step 0's y fragments into staging buffer 0 via the TDM.
    if (ht == 0) {
        unsigned long long ga = (unsigned long long)yP;   // t=0 block
        u32x4 g0;
        g0[0] = 1u;                                       // count=1 (valid), user mode
        g0[1] = ldsBaseY;                                 // lds_addr
        g0[2] = (unsigned)(ga & 0xFFFFFFFFu);             // global_addr[31:0]
        g0[3] = (unsigned)((ga >> 32) & 0x01FFFFFFu) | (2u << 30); // addr[56:32] | type=2
        i32x8 g1;
        g1[0] = (3 << 16);                                // data_size=3 (8B), no multicast
        g1[1] = (1024 << 16);                             // tensor_dim0[15:0] = 1024
        g1[2] = (1 << 16);                                // tensor_dim0 hi=0, tensor_dim1 lo=1
        g1[3] = (1024 << 16);                             // tensor_dim1 hi=0, tile_dim0=1024
        g1[4] = 1;                                        // tile_dim1=1, tile_dim2=0
        g1[5] = 1024; g1[6] = 0; g1[7] = 0;               // dim0_stride=1024
        i32x4 gz4 = {0, 0, 0, 0};
        i32x8 gz8 = {0, 0, 0, 0, 0, 0, 0, 0};
        __builtin_amdgcn_tensor_load_to_lds(g0, g1, gz4, gz4, gz8, 0);
        __builtin_amdgcn_s_wait_tensorcnt(0);
    }
#endif

    __syncthreads();

    for (int t = 0; t < T_; ++t) {
        unsigned short* hRead  = hA + (t & 1) * 32768;
        unsigned short* hWrite = hA + ((t + 1) & 1) * 32768;

#if USE_TDM
        const unsigned short* yCur = yS + (t & 1) * 4096;   // 8 KB staged fragments
        // kick off DMA of next step's y block into the other staging buffer
        if (ht == 0 && t + 1 < T_) {
            unsigned long long ga =
                (unsigned long long)yP + (unsigned long long)(t + 1) * 8192ull;
            u32x4 g0;
            g0[0] = 1u;
            g0[1] = ldsBaseY + (unsigned)(((t + 1) & 1) * 8192);
            g0[2] = (unsigned)(ga & 0xFFFFFFFFu);
            g0[3] = (unsigned)((ga >> 32) & 0x01FFFFFFu) | (2u << 30);
            i32x8 g1;
            g1[0] = (3 << 16);
            g1[1] = (1024 << 16);
            g1[2] = (1 << 16);
            g1[3] = (1024 << 16);
            g1[4] = 1;
            g1[5] = 1024; g1[6] = 0; g1[7] = 0;
            i32x4 gz4 = {0, 0, 0, 0};
            i32x8 gz8 = {0, 0, 0, 0, 0, 0, 0, 0};
            __builtin_amdgcn_tensor_load_to_lds(g0, g1, gz4, gz4, gz8, 0);
        }
#else
        if (t + 1 < T_)
            __builtin_prefetch((const void*)(yP + (((size_t)(t + 1) * 8) * 32 + lane) * 16), 0, 0);
#endif

        for (int p = 0; p < 2; ++p) {          // two m-passes of 4 batch tiles
            // Launder a zero offset PER m-pass: weight-fragment addresses are
            // variant w.r.t. both loops -> loads stay in the innermost body as
            // global_load_b128 (no LICM -> no spill, no flat-alias penalty).
            unsigned zoff = 0;
            asm volatile("" : "+s"(zoff));
            const unsigned short* wp = whhP + zoff;

            v8f acc[4][4];

            // ---- input contribution (K=32): starts the chain with inline C=0 ----
#pragma unroll
            for (int m = 0; m < 4; ++m) {
                const int mt = p * 4 + m;
#if USE_TDM
                const v16bf Ay = *(const v16bf*)(yCur + (mt * 32 + lane) * 16);
#else
                const v16bf Ay =
                    *(const v16bf*)(yP + (((size_t)t * 8 + mt) * 32 + lane) * 16);
#endif
#pragma unroll
                for (int g = 0; g < 4; ++g) {
                    v8f z = {0.f, 0.f, 0.f, 0.f, 0.f, 0.f, 0.f, 0.f};
                    acc[m][g] = __builtin_amdgcn_wmma_f32_16x16x32_bf16(
                        false, Ay, false, Bi[g], (short)0, z, false, false);
                }
            }

            // ---- recurrent contribution: K = 256 (8 k-tiles) ----
#pragma unroll
            for (int kt = 0; kt < 8; ++kt) {
                v16bf Bg[4];
#pragma unroll
                for (int g = 0; g < 4; ++g)
                    Bg[g] = *(const v16bf*)(wp + ((((ht * 4 + g) * 8 + kt) * 32) + lane) * 16);
#pragma unroll
                for (int m = 0; m < 4; ++m) {
                    const int mt = p * 4 + m;
                    const v16bf Ah =
                        *(const v16bf*)&hRead[((mt * 8 + kt) * 32 + lane) * 16];
#pragma unroll
                    for (int g = 0; g < 4; ++g)
                        acc[m][g] = __builtin_amdgcn_wmma_f32_16x16x32_bf16(
                            false, Ah, false, Bg[g], (short)0, acc[m][g], false, false);
                }
            }

            // ---- gate fusion: c in LDS, h_new scattered into NEXT h buffer ----
#pragma unroll
            for (int m = 0; m < 4; ++m) {
                const int mt = p * 4 + m;
#pragma unroll
                for (int r = 0; r < 8; ++r) {
                    const int b = mt * 16 + r + off;
                    float cv = cS[b * H_ + hidx];
                    float gi = sigmoidf_(acc[m][0][r] + bias[0]);
                    float gf = sigmoidf_(acc[m][1][r] + bias[1]);
                    float gg = tanhf(acc[m][2][r] + bias[2]);
                    float go = sigmoidf_(acc[m][3][r] + bias[3]);
                    cv = gf * cv + gi * gg;
                    cS[b * H_ + hidx] = cv;
                    float h = go * tanhf(cv);
                    hWrite[((mt * 8 + ktp) * 32 + (r + off) + 16 * hi16) * 16 + vv * 2 + xx] =
                        f2bf(h);
                    if (t == T_ - 1) hFin[b * H_ + hidx] = h;
                }
            }
        }

#if USE_TDM
        if (ht == 0) __builtin_amdgcn_s_wait_tensorcnt(0);  // next y block landed
#endif
        __syncthreads();   // h_new + c + staged y visible before next step
    }
}

// ---------------------------------------------------------------------------
// Tiny MLP head: out = x + fc2( relu( fc1( [x ; h] ) ) )   (~19 MFLOP)
// ---------------------------------------------------------------------------
__global__ __launch_bounds__(256)
void mlp_head(const float* __restrict__ x_b, const float* __restrict__ hF,
              const float* __restrict__ fc1w, const float* __restrict__ fc1b,
              const float* __restrict__ fc2w, const float* __restrict__ fc2b,
              float* __restrict__ out) {
    const int b = blockIdx.x;
    const int o = threadIdx.x;
    __shared__ float inp[Y_ + H_];
    __shared__ float act[H_];

    if (o < Y_) inp[o] = x_b[b * Y_ + o];
    inp[Y_ + o] = hF[b * H_ + o];
    __syncthreads();

    float s = fc1b[o];
    const float* w = fc1w + o * (Y_ + H_);
#pragma unroll 8
    for (int j = 0; j < Y_ + H_; ++j) s = fmaf(w[j], inp[j], s);
    act[o] = fmaxf(s, 0.f);
    __syncthreads();

    if (o < Y_) {
        float s2 = fc2b[o];
        const float* w2 = fc2w + o * H_;
#pragma unroll 8
        for (int j = 0; j < H_; ++j) s2 = fmaf(w2[j], act[j], s2);
        out[b * Y_ + o] = x_b[b * Y_ + o] + s2;
    }
}

// ---------------------------------------------------------------------------
extern "C" void kernel_launch(void* const* d_in, const int* in_sizes, int n_in,
                              void* d_out, int out_size, void* d_ws, size_t ws_size,
                              hipStream_t stream) {
    const float* x_b   = (const float*)d_in[0];
    const float* y_seq = (const float*)d_in[1];
    const float* W_ih  = (const float*)d_in[2];
    const float* W_hh  = (const float*)d_in[3];
    const float* b_ih  = (const float*)d_in[4];
    const float* b_hh  = (const float*)d_in[5];
    const float* fc1w  = (const float*)d_in[6];
    const float* fc1b  = (const float*)d_in[7];
    const float* fc2w  = (const float*)d_in[8];
    const float* fc2b  = (const float*)d_in[9];
    float* out = (float*)d_out;

    char* ws = (char*)d_ws;
    unsigned short* whhP = (unsigned short*)ws;                              // 512 KB
    unsigned short* wihP = (unsigned short*)(ws + (512u << 10));             // 64 KB
    float*          hFin = (float*)(ws + (512u << 10) + (64u << 10));        // 128 KB
    unsigned short* yP   = (unsigned short*)(ws + (512u << 10) + (64u << 10) + (128u << 10)); // 8 MB

    pack_whh<<<(16 * 4 * 8 * 512 + 255) / 256, 256, 0, stream>>>(W_hh, whhP);
    pack_wih<<<(4 * 16 * 512 + 255) / 256, 256, 0, stream>>>(W_ih, wihP);
    pack_y  <<<(T_ * 8 * 512 + 255) / 256, 256, 0, stream>>>(y_seq, yP);

    // dynamic LDS: 2x64 KB h double-buffer + 128 KB c-state + 16 KB y staging = 272 KB
    lstm_persistent<<<1, 512, 278528, stream>>>(whhP, wihP, yP, b_ih, b_hh, hFin);

    mlp_head<<<B_, 256, 0, stream>>>(x_b, hFin, fc1w, fc1b, fc2w, fc2b, out);
}
